// SparsemaxAttention_8143257993564
// MI455X (gfx1250) — matrix operand order repaired and checked by reading, not compiled, not run on hardware
//
#include <hip/hip_runtime.h>
#include <hip/hip_bf16.h>

typedef __attribute__((ext_vector_type(16))) _Float16 v16h;
typedef __attribute__((ext_vector_type(8)))  _Float16 v8h;
typedef __attribute__((ext_vector_type(8)))  float    v8f;
typedef __attribute__((ext_vector_type(4)))  int      v4i;

typedef __attribute__((address_space(1))) v4i* gv4i_p;   // global (AS1)
typedef __attribute__((address_space(3))) v4i* lv4i_p;   // LDS (AS3)

#define WMMA_F16(A, B, C) \
    __builtin_amdgcn_wmma_f32_16x16x32_f16(false, (A), false, (B), (short)0, (C), false, false)

#if defined(__has_builtin)
#if __has_builtin(__builtin_amdgcn_global_load_async_to_lds_b128)
#define HAVE_ASYNC_LDS 1
#endif
#endif

__device__ inline void wait_asynccnt0() {
#if defined(__has_builtin) && __has_builtin(__builtin_amdgcn_s_wait_asynccnt)
    __builtin_amdgcn_s_wait_asynccnt(0);
#else
    asm volatile("s_wait_asynccnt 0x0" ::: "memory");
#endif
}

// ---------------------------------------------------------------------------
// Kernel 0: convert + transpose weights to f16 [N][K] so WMMA B-operands are
// contiguous 32B per lane.
// ---------------------------------------------------------------------------
__global__ __launch_bounds__(256) void prep_weights(
    const float* __restrict__ wa,   // [1024][3072]
    const float* __restrict__ wp,   // [1024][1024]
    _Float16* __restrict__ waT,     // [3072][1024]
    _Float16* __restrict__ wpT)     // [1024][1024]
{
    size_t e = (size_t)blockIdx.x * 256 + threadIdx.x;
    const size_t NA = (size_t)3072 * 1024;
    if (e < NA) {
        size_t k = e / 3072, n = e % 3072;          // coalesced source read
        waT[n * 1024 + k] = (_Float16)wa[e];
    } else {
        size_t e2 = e - NA;
        if (e2 < (size_t)1024 * 1024) {
            size_t k = e2 >> 10, n = e2 & 1023;
            wpT[n * 1024 + k] = (_Float16)wp[e2];
        }
    }
}

// A-fragment (16x32 f16, contiguous k = 16h + s layout) built from an f32 row
__device__ inline v16h afrag_f32(const float* __restrict__ row, int o) {
    v16h a;
#pragma unroll
    for (int i = 0; i < 8; ++i) a[i]     = (_Float16)row[o + i];
#pragma unroll
    for (int i = 0; i < 8; ++i) a[8 + i] = (_Float16)row[o + 8 + i];
    return a;
}
// A-fragment from an f16 row (two contiguous 16B loads)
__device__ inline v16h afrag_f16(const _Float16* __restrict__ row, int o) {
    v8h t0 = *(const v8h*)(row + o);
    v8h t1 = *(const v8h*)(row + o + 8);
    v16h a;
#pragma unroll
    for (int i = 0; i < 8; ++i) { a[i] = t0[i]; a[8 + i] = t1[i]; }
    return a;
}

// ---------------------------------------------------------------------------
// Kernel 1: qkv = x @ w_attn + b_attn ; scatter to q[B,H,T,hd], k[B,H,T,hd],
// vT[B,H,hd,T], all f16.  Block = 8 waves; wave tile = 32x64 (2 M-tiles so
// every B fragment feeds two WMMAs).  Block tile = 256x64.
// ---------------------------------------------------------------------------
__global__ __launch_bounds__(256) void qkv_gemm(
    const float* __restrict__ x,      // [4096][1024]
    const _Float16* __restrict__ wT,  // [3072][1024]
    const float* __restrict__ bias,   // [3072]
    _Float16* __restrict__ qb, _Float16* __restrict__ kb, _Float16* __restrict__ vT)
{
    const int lane = threadIdx.x & 31;
    const int wave = threadIdx.x >> 5;
    const bool hi  = lane >= 16;
    const int l16  = lane & 15;

    const int row0  = blockIdx.x * 256 + wave * 32 + l16; // M-tile 0 row for this lane
    const int nbase = blockIdx.y * 64;

    const float* xrow0 = x + (size_t)row0 * 1024;
    const float* xrow1 = xrow0 + 16 * 1024;

    v8f acc[2][4];
#pragma unroll
    for (int mt = 0; mt < 2; ++mt)
#pragma unroll
        for (int nt = 0; nt < 4; ++nt)
            acc[mt][nt] = (v8f){0.f,0.f,0.f,0.f,0.f,0.f,0.f,0.f};

    for (int kk = 0; kk < 1024; kk += 32) {
        const int o = kk + (hi ? 16 : 0);
        if (kk + 256 < 1024) {                   // stream-ahead hint
            __builtin_prefetch(xrow0 + kk + 256, 0, 0);
            __builtin_prefetch(xrow1 + kk + 256, 0, 0);
        }
        v16h a0 = afrag_f32(xrow0, o);
        v16h a1 = afrag_f32(xrow1, o);

#pragma unroll
        for (int nt = 0; nt < 4; ++nt) {
            const _Float16* bp =
                wT + (size_t)(nbase + nt * 16 + l16) * 1024 + kk + (hi ? 16 : 0);
            v16h bf = *(const v16h*)bp;
            acc[0][nt] = WMMA_F16(a0, bf, acc[0][nt]);
            acc[1][nt] = WMMA_F16(a1, bf, acc[1][nt]);
        }
    }

    // per-lane bias values (only 4 distinct: one per nt tile) -- hoisted
    float bv[4];
#pragma unroll
    for (int nt = 0; nt < 4; ++nt) bv[nt] = bias[nbase + nt * 16 + l16];

    // ---- scatter.  which & h are BLOCK-UNIFORM (nbase % 64 == 0, 64 | 1024) ----
    const int which = nbase >> 10;          // 0=q 1=k 2=v
    const int h     = (nbase & 1023) >> 6;  // head
    // C/D layout: vgpr j, lane<16 -> (M=j,N=lane), lane>=16 -> (M=j+8,N=lane-16)
    if (which < 2) {
        _Float16* dst = which ? kb : qb;
#pragma unroll
        for (int mt = 0; mt < 2; ++mt)
#pragma unroll
            for (int nt = 0; nt < 4; ++nt)
#pragma unroll
                for (int j = 0; j < 8; ++j) {
                    const int m    = j + (hi ? 8 : 0);
                    const int grow = blockIdx.x * 256 + wave * 32 + mt * 16 + m;
                    const int d    = nt * 16 + l16;
                    const float v  = acc[mt][nt][j] + bv[nt];
                    const int b = grow >> 11, t = grow & 2047;
                    dst[(((size_t)b * 16 + h) * 2048 + t) * 64 + d] = (_Float16)v;
                }
    } else {
#pragma unroll
        for (int mt = 0; mt < 2; ++mt)
#pragma unroll
            for (int nt = 0; nt < 4; ++nt)
#pragma unroll
                for (int j = 0; j < 8; ++j) {
                    const int m    = j + (hi ? 8 : 0);
                    const int grow = blockIdx.x * 256 + wave * 32 + mt * 16 + m;
                    const int d    = nt * 16 + l16;
                    const float v  = acc[mt][nt][j] + bv[nt];
                    const int b = grow >> 11, t = grow & 2047;
                    vT[(((size_t)b * 16 + h) * 64 + d) * 2048 + t] = (_Float16)v;
                }
    }
}

// ---------------------------------------------------------------------------
// Kernel 2: causal sparsemax attention for one (b, h, 16-query tile).
// Scores for the whole causal row in LDS (16 x 2048 f32 = 128KB; WGP has
// 320KB).  K tiles staged into LDS with gfx1250 async loads (ASYNCcnt).
// ---------------------------------------------------------------------------
__global__ __launch_bounds__(256) void attn_sparsemax(
    const _Float16* __restrict__ qb,  // [B,H,T,hd]
    const _Float16* __restrict__ kb,  // [B,H,T,hd]
    const _Float16* __restrict__ vT,  // [B,H,hd,T]
    _Float16* __restrict__ y)         // [B,T,C] f16
{
    __shared__ float sc[16 * 2048];
    __shared__ float yacc[16 * 64];
    __shared__ __align__(16) _Float16 kstage[8][16 * 64];  // 2KB per wave

    const int qt = blockIdx.x, h = blockIdx.y, b = blockIdx.z;
    const int lane = threadIdx.x & 31;
    const int wave = threadIdx.x >> 5;
    const bool hiL = lane >= 16;
    const int l16  = lane & 15;

    const int qbase = qt * 16;
    const int L     = qbase + 16;
    const int nkt   = qt + 1;

    const size_t hd_off = (size_t)b * 16 + h;
    const _Float16* qh = qb + hd_off * 2048 * 64;
    const _Float16* kh = kb + hd_off * 2048 * 64;
    const _Float16* vh = vT + hd_off * 64 * 2048;

    // ---- q A-frags (hd = 64 -> two K=32 chunks) ----
    const _Float16* qrow = qh + (size_t)(qbase + l16) * 64;
    v16h a_lo = afrag_f16(qrow, hiL ? 16 : 0);
    v16h a_hi = afrag_f16(qrow + 32, hiL ? 16 : 0);

    // ---- Phase A: S = q @ k^T / 8, causal mask, into LDS ----
    _Float16* stage = &kstage[wave][0];
    for (int kt = wave; kt < nkt; kt += 8) {
        const _Float16* ksrc = kh + (size_t)kt * 16 * 64;   // 16 rows x 64 halfs = 2KB
#ifdef HAVE_ASYNC_LDS
#pragma unroll
        for (int c = 0; c < 4; ++c) {                        // 4 x (32 lanes x 16B)
            __builtin_amdgcn_global_load_async_to_lds_b128(
                (gv4i_p)(void*)((char*)ksrc + c * 512 + lane * 16),
                (lv4i_p)(void*)((char*)stage + c * 512 + lane * 16),
                0, 0);
        }
        wait_asynccnt0();
#else
#pragma unroll
        for (int c = 0; c < 4; ++c)
            *(v8h*)((char*)stage + c * 512 + lane * 16) =
                *(const v8h*)((const char*)ksrc + c * 512 + lane * 16);
#endif
        const _Float16* krow = stage + l16 * 64 + (hiL ? 16 : 0);
        v16h b0 = *(const v16h*)(krow);        // dims 0..31
        v16h b1 = *(const v16h*)(krow + 32);   // dims 32..63
        v8f s = (v8f){0.f,0.f,0.f,0.f,0.f,0.f,0.f,0.f};
        s = WMMA_F16(a_lo, b0, s);
        s = WMMA_F16(a_hi, b1, s);
        const int key = kt * 16 + l16;
#pragma unroll
        for (int j = 0; j < 8; ++j) {
            const int m = j + (hiL ? 8 : 0);
            float v = s[j] * 0.125f;                     // 1/sqrt(64)
            if (key > qbase + m) v = -__builtin_inff();  // causal
            sc[m * 2048 + key] = v;
        }
    }
    __syncthreads();

    // ---- Phase B: exact sparsemax per row via bisection on tau ----
    for (int r = wave * 2; r < wave * 2 + 2; ++r) {
        float* zr = sc + r * 2048;
        float mx = -__builtin_inff();
        for (int i = lane; i < L; i += 32) mx = fmaxf(mx, zr[i]);
#pragma unroll
        for (int off = 16; off > 0; off >>= 1) mx = fmaxf(mx, __shfl_xor(mx, off, 32));

        float lo = mx - 1.0f, hiT = mx;   // f(lo) >= 1, f(hi) = 0
#pragma unroll 1
        for (int it = 0; it < 30; ++it) {
            const float tau = 0.5f * (lo + hiT);
            float s = 0.0f;
            for (int i = lane; i < L; i += 32) s += fmaxf(zr[i] - tau, 0.0f);
#pragma unroll
            for (int off = 16; off > 0; off >>= 1) s += __shfl_xor(s, off, 32);
            if (s > 1.0f) lo = tau; else hiT = tau;
        }
        const float tau = 0.5f * (lo + hiT);
        float s = 0.0f;
        for (int i = lane; i < L; i += 32) s += fmaxf(zr[i] - tau, 0.0f);
#pragma unroll
        for (int off = 16; off > 0; off >>= 1) s += __shfl_xor(s, off, 32);
        const float inv = 1.0f / s;
        for (int i = lane; i < L; i += 32) zr[i] = fmaxf(zr[i] - tau, 0.0f) * inv;
    }
    __syncthreads();

    // ---- Phase C: y = P @ V ; nt = hd tile (4 waves), halfk = K split (2) ----
    const int nt = wave & 3, halfk = wave >> 2;
    const int nc = (L + 31) >> 5;
    const int d0 = nt * 16 + l16;
    const _Float16* vrow = vh + (size_t)d0 * 2048 + (hiL ? 16 : 0);

    v8f acc = (v8f){0.f,0.f,0.f,0.f,0.f,0.f,0.f,0.f};
    for (int c = halfk; c < nc; c += 2) {
        v16h pa;
#pragma unroll
        for (int e = 0; e < 16; ++e) {
            const int kof = e + (hiL ? 16 : 0);
            const int col = c * 32 + kof;
            const float v = (col < L) ? sc[l16 * 2048 + col] : 0.0f;
            pa[e] = (_Float16)v;
        }
        v16h bv = *(const v16h*)(vrow + c * 32);
        acc = WMMA_F16(pa, bv, acc);
    }

    if (halfk == 0) {
#pragma unroll
        for (int j = 0; j < 8; ++j) yacc[(j + (hiL ? 8 : 0)) * 64 + d0] = acc[j];
    }
    __syncthreads();
    if (halfk == 1) {
#pragma unroll
        for (int j = 0; j < 8; ++j) yacc[(j + (hiL ? 8 : 0)) * 64 + d0] += acc[j];
    }
    __syncthreads();

    for (int e = threadIdx.x; e < 1024; e += 256) {
        const int m = e >> 6, d = e & 63;
        y[((size_t)b * 2048 + qbase + m) * 1024 + h * 64 + d] = (_Float16)yacc[e];
    }
}

// ---------------------------------------------------------------------------
// Kernel 3: out = y @ w_proj + b_proj  (f16 x f16 -> f32); 2 M-tiles / wave.
// ---------------------------------------------------------------------------
__global__ __launch_bounds__(256) void proj_gemm(
    const _Float16* __restrict__ y,   // [4096][1024]
    const _Float16* __restrict__ wT,  // [1024][1024]
    const float* __restrict__ bias,   // [1024]
    float* __restrict__ out)          // [4096][1024]
{
    const int lane = threadIdx.x & 31;
    const int wave = threadIdx.x >> 5;
    const bool hi  = lane >= 16;
    const int l16  = lane & 15;

    const int row0  = blockIdx.x * 256 + wave * 32 + l16;
    const int nbase = blockIdx.y * 64;
    const _Float16* yrow0 = y + (size_t)row0 * 1024;
    const _Float16* yrow1 = yrow0 + 16 * 1024;

    v8f acc[2][4];
#pragma unroll
    for (int mt = 0; mt < 2; ++mt)
#pragma unroll
        for (int nt = 0; nt < 4; ++nt)
            acc[mt][nt] = (v8f){0.f,0.f,0.f,0.f,0.f,0.f,0.f,0.f};

    for (int kk = 0; kk < 1024; kk += 32) {
        const int o = kk + (hi ? 16 : 0);
        if (kk + 256 < 1024) {
            __builtin_prefetch(yrow0 + kk + 256, 0, 0);
            __builtin_prefetch(yrow1 + kk + 256, 0, 0);
        }
        v16h a0 = afrag_f16(yrow0, o);
        v16h a1 = afrag_f16(yrow1, o);

#pragma unroll
        for (int nt = 0; nt < 4; ++nt) {
            const _Float16* bp =
                wT + (size_t)(nbase + nt * 16 + l16) * 1024 + kk + (hi ? 16 : 0);
            v16h bf = *(const v16h*)bp;
            acc[0][nt] = WMMA_F16(a0, bf, acc[0][nt]);
            acc[1][nt] = WMMA_F16(a1, bf, acc[1][nt]);
        }
    }

    float bv[4];
#pragma unroll
    for (int nt = 0; nt < 4; ++nt) bv[nt] = bias[nbase + nt * 16 + l16];

#pragma unroll
    for (int mt = 0; mt < 2; ++mt)
#pragma unroll
        for (int nt = 0; nt < 4; ++nt)
#pragma unroll
            for (int j = 0; j < 8; ++j) {
                const int m    = j + (hi ? 8 : 0);
                const int grow = blockIdx.x * 256 + wave * 32 + mt * 16 + m;
                const int n    = nbase + nt * 16 + l16;
                out[(size_t)grow * 1024 + n] = acc[mt][nt][j] + bv[nt];
            }
}

// ---------------------------------------------------------------------------
extern "C" void kernel_launch(void* const* d_in, const int* in_sizes, int n_in,
                              void* d_out, int out_size, void* d_ws, size_t ws_size,
                              hipStream_t stream) {
    const float* x      = (const float*)d_in[0];
    const float* w_attn = (const float*)d_in[1];
    const float* b_attn = (const float*)d_in[2];
    const float* w_proj = (const float*)d_in[3];
    const float* b_proj = (const float*)d_in[4];
    float* out = (float*)d_out;

    char* ws = (char*)d_ws;
    _Float16* waT = (_Float16*)(ws);                      // 6 MB
    _Float16* wpT = (_Float16*)(ws + (6u  << 20));        // 2 MB
    _Float16* qb  = (_Float16*)(ws + (8u  << 20));        // 8 MB
    _Float16* kb  = (_Float16*)(ws + (16u << 20));        // 8 MB
    _Float16* vT  = (_Float16*)(ws + (24u << 20));        // 8 MB
    _Float16* yb  = (_Float16*)(ws + (32u << 20));        // 8 MB

    prep_weights<<<16384, 256, 0, stream>>>(w_attn, w_proj, waT, wpT);
    qkv_gemm<<<dim3(16, 48), 256, 0, stream>>>(x, waT, b_attn, qb, kb, vT);
    attn_sparsemax<<<dim3(128, 16, 2), 256, 0, stream>>>(qb, kb, vT, yb);
    proj_gemm<<<dim3(16, 16), 256, 0, stream>>>(yb, wpT, b_proj, out);
}
